// MoEModel_81149112091138
// MI455X (gfx1250) — matrix operand order repaired and checked by reading, not compiled, not run on hardware
//
#include <hip/hip_runtime.h>
#include <cstdint>

// ---------------- types ----------------
typedef __attribute__((ext_vector_type(16))) __bf16 v16bf;
typedef __attribute__((ext_vector_type(8)))  __bf16 v8bf;
typedef __attribute__((ext_vector_type(8)))  float  v8f;

// ---------------- problem constants ----------------
#define NTOK   16384        // B*S = 8*2048
#define HD     512          // hidden
#define ED     8            // experts
#define FH     2048         // 4*H
#define MAXT   16384        // max tokens per expert bucket
#define TM     32           // tokens per tile (2 x 16-row WMMA tiles)
#define MAX_TILES (MAXT/TM) // 512
#define FCH    128          // f-chunk per outer iteration (8 waves * 16 cols)

// ---------------- fragment helpers ----------------
// A fragment (16x32 bf16, M=lane%16): lane half 0 holds
// K={k0..k0+15}, half 1 holds K={k0+16..k0+31} (matches frag_b).
__device__ __forceinline__ v16bf frag_a(const __bf16* row, int k0, int half) {
    v8bf lo = *(const v8bf*)(row + k0 + half * 16);
    v8bf hi = *(const v8bf*)(row + k0 + half * 16 + 8);
    return __builtin_shufflevector(lo, hi, 0,1,2,3,4,5,6,7,8,9,10,11,12,13,14,15);
}
// B fragment (32x16 bf16, N=lane%16): lane holds 16 contiguous K values
// (lanes 0-15: K=k0..k0+15, lanes 16-31: K=k0+16..k0+31) of column N.
// Weights are pre-transposed so this is one contiguous 32-byte run.
__device__ __forceinline__ v16bf frag_b(const __bf16* p) {
    return *(const v16bf*)p;
}

__device__ __forceinline__ float gelu_exact(float v) {
    return 0.5f * v * (1.0f + erff(v * 0.70710678118654752f));
}

// ---------------- kernel 1: fp32 -> bf16 elementwise ----------------
__global__ void cvt_bf16_kernel(const float* __restrict__ in, __bf16* __restrict__ out, int n) {
    int i = blockIdx.x * blockDim.x + threadIdx.x;
    if (i < n) out[i] = (__bf16)in[i];
}

// ---------------- kernel 2: fp32 [E][R][C] -> bf16 [E][C][R] (tiled transpose) ----------------
__global__ __launch_bounds__(256) void transpose_cvt_kernel(const float* __restrict__ in,
                                                            __bf16* __restrict__ out,
                                                            int R, int C) {
    __shared__ float t[32][33];
    const int e  = blockIdx.z;
    const int c0 = blockIdx.x * 32;
    const int r0 = blockIdx.y * 32;
    const int tx = threadIdx.x & 31;
    const int ty = threadIdx.x >> 5;          // 0..7
    const float*  src = in  + (size_t)e * R * C;
    __bf16*       dst = out + (size_t)e * R * C;
    #pragma unroll
    for (int i = 0; i < 4; ++i) {
        int r = ty + i * 8;
        t[r][tx] = src[(size_t)(r0 + r) * C + c0 + tx];
    }
    __syncthreads();
    #pragma unroll
    for (int i = 0; i < 4; ++i) {
        int c = ty + i * 8;
        dst[(size_t)(c0 + c) * R + r0 + tx] = (__bf16)t[tx][c];
    }
}

// ---------------- kernel 3: gating (softmax top-2 + bucket build) ----------------
__global__ void gating_kernel(const float* __restrict__ x, const float* __restrict__ Wg,
                              int* __restrict__ counts, int* __restrict__ tok_list,
                              float* __restrict__ wgt_list) {
    int t = blockIdx.x * blockDim.x + threadIdx.x;
    if (t >= NTOK) return;
    float l[ED];
    #pragma unroll
    for (int e = 0; e < ED; ++e) l[e] = 0.0f;
    const float* xr = x + (size_t)t * HD;
    for (int h = 0; h < HD; ++h) {
        float xv = xr[h];
        const float* wr = Wg + h * ED;
        #pragma unroll
        for (int e = 0; e < ED; ++e) l[e] += xv * wr[e];
    }
    float m = l[0];
    #pragma unroll
    for (int e = 1; e < ED; ++e) m = fmaxf(m, l[e]);
    float p[ED], s = 0.0f;
    #pragma unroll
    for (int e = 0; e < ED; ++e) { p[e] = __expf(l[e] - m); s += p[e]; }
    float inv = 1.0f / s;
    #pragma unroll
    for (int e = 0; e < ED; ++e) p[e] *= inv;
    // top-2, first-occurrence on ties (matches top_k ordering)
    int i1 = 0;
    #pragma unroll
    for (int e = 1; e < ED; ++e) if (p[e] > p[i1]) i1 = e;
    int i2 = (i1 == 0) ? 1 : 0;
    #pragma unroll
    for (int e = 0; e < ED; ++e) if (e != i1 && p[e] > p[i2]) i2 = e;

    int pos1 = atomicAdd(&counts[i1], 1);
    tok_list[i1 * MAXT + pos1] = t;
    wgt_list[i1 * MAXT + pos1] = p[i1];
    int pos2 = atomicAdd(&counts[i2], 1);
    tok_list[i2 * MAXT + pos2] = t;
    wgt_list[i2 * MAXT + pos2] = p[i2];
}

// ---------------- kernel 4: fused expert FFN (WMMA bf16, 32-token tiles) ----------------
// grid: (MAX_TILES, E), block: 256 (8 waves). One block = one 32-token tile.
// Each B-fragment (1KB) feeds TWO wmma ops (both 16-row halves of the tile),
// halving L2 B-traffic vs a 16-token tile.
__global__ __launch_bounds__(256) void moe_expert_kernel(
    const __bf16* __restrict__ xbf,    // [NTOK][HD]
    const __bf16* __restrict__ w1t,    // [E][FH][HD]   (f-major, k contiguous)
    const __bf16* __restrict__ w2t,    // [E][HD][FH]   (h-major, f contiguous)
    const float*  __restrict__ b1,     // [E][FH]
    const float*  __restrict__ b2,     // [E][HD]
    const int*    __restrict__ counts,
    const int*    __restrict__ tok_list,
    const float*  __restrict__ wgt_list,
    float* __restrict__ out)           // [NTOK][HD]
{
    __shared__ __bf16 sX[TM][HD + 8];    // gathered token rows (bf16), ~33 KB
    __shared__ __bf16 sH[TM][FCH + 8];   // gelu(h) chunk (bf16), ~8.7 KB
    __shared__ int    sTok[TM];
    __shared__ float  sWgt[TM];

    const int e    = blockIdx.y;
    const int tile = blockIdx.x;
    const int ne   = counts[e];
    if (tile * TM >= ne) return;

    const int tid  = threadIdx.x;
    const int lane = tid & 31;
    const int wave = tid >> 5;
    const int half = lane >> 4;
    const int l16  = lane & 15;

    if (tid < TM) {
        int idx = tile * TM + tid;
        if (idx < ne) {
            sTok[tid] = tok_list[e * MAXT + idx];
            sWgt[tid] = wgt_list[e * MAXT + idx];
        } else {                       // padded row: duplicate row 0, zero weight
            sTok[tid] = tok_list[e * MAXT];
            sWgt[tid] = 0.0f;
        }
    }
    __syncthreads();

    // ---- gather TM token rows into LDS via CDNA5 async load-to-LDS ----
    {
        const int row = tid >> 3;                 // 8 threads per row
        const int c0  = (tid & 7) * 64;           // 64 bf16 = 128B per thread
        const uint32_t gBase   = ((uint32_t)sTok[row] * (uint32_t)HD + (uint32_t)c0) * 2u;
        const uint32_t ldsBase = (uint32_t)(uintptr_t)(&sX[row][c0]);
        #pragma unroll
        for (int j = 0; j < 8; ++j) {
            uint32_t lo = ldsBase + (uint32_t)(j * 16);
            uint32_t go = gBase   + (uint32_t)(j * 16);
            asm volatile("global_load_async_to_lds_b128 %0, %1, %2"
                         :: "v"(lo), "v"(go), "s"(xbf) : "memory");
        }
    }
    asm volatile("s_wait_asynccnt 0" ::: "memory");
    __syncthreads();

    // ---- persistent output accumulators: this wave owns cols [wave*64, wave*64+64)
    //      acc[nt][rt]: nt = 16-col tile (4), rt = 16-row half of the token tile (2)
    v8f acc[4][2];
    #pragma unroll
    for (int nt = 0; nt < 4; ++nt) {
        acc[nt][0] = v8f{};
        acc[nt][1] = v8f{};
    }

    for (int fc = 0; fc < FH; fc += FCH) {
        // stage 1: Hc[:, wave*16 .. +16) = gelu(X @ W1 + b1)
        const int ncol = fc + wave * 16 + l16;                 // global f column per lane
        const __bf16* w1p = w1t + ((size_t)e * FH + (size_t)ncol) * HD + half * 16;
        v8f hacc0 = v8f{};
        v8f hacc1 = v8f{};
        #pragma unroll 4
        for (int k0 = 0; k0 < HD; k0 += 32) {
            v16bf b  = frag_b(w1p + k0);
            v16bf a0 = frag_a(&sX[l16][0],      k0, half);
            v16bf a1 = frag_a(&sX[l16 + 16][0], k0, half);
            hacc0 = __builtin_amdgcn_wmma_f32_16x16x32_bf16(
                false, a0, false, b, (short)0, hacc0, false, false);
            hacc1 = __builtin_amdgcn_wmma_f32_16x16x32_bf16(
                false, a1, false, b, (short)0, hacc1, false, false);
        }
        {
            const float bias = b1[e * FH + ncol];
            const int   ccol = wave * 16 + l16;                // chunk-local column
            #pragma unroll
            for (int r = 0; r < 8; ++r) {
                sH[r + 8 * half][ccol]      = (__bf16)gelu_exact(hacc0[r] + bias);
                sH[16 + r + 8 * half][ccol] = (__bf16)gelu_exact(hacc1[r] + bias);
            }
        }
        __syncthreads();

        // stage 2: acc += Hc @ W2[fc:fc+FCH, :]
        #pragma unroll
        for (int nt = 0; nt < 4; ++nt) {
            const int hcol = wave * 64 + nt * 16 + l16;        // output column per lane
            const __bf16* w2p = w2t + ((size_t)e * HD + (size_t)hcol) * FH + fc + half * 16;
            #pragma unroll
            for (int k0 = 0; k0 < FCH; k0 += 32) {
                v16bf b  = frag_b(w2p + k0);
                v16bf a0 = frag_a(&sH[l16][0],      k0, half);
                v16bf a1 = frag_a(&sH[l16 + 16][0], k0, half);
                acc[nt][0] = __builtin_amdgcn_wmma_f32_16x16x32_bf16(
                    false, a0, false, b, (short)0, acc[nt][0], false, false);
                acc[nt][1] = __builtin_amdgcn_wmma_f32_16x16x32_bf16(
                    false, a1, false, b, (short)0, acc[nt][1], false, false);
            }
        }
        __syncthreads();   // before next iteration overwrites sH
    }

    // ---- epilogue: +b2, * gate weight, scatter-add ----
    #pragma unroll
    for (int nt = 0; nt < 4; ++nt) {
        const int hcol = wave * 64 + nt * 16 + l16;
        const float bias = b2[e * HD + hcol];
        #pragma unroll
        for (int rt = 0; rt < 2; ++rt) {
            #pragma unroll
            for (int r = 0; r < 8; ++r) {
                const int row = rt * 16 + r + 8 * half;
                float val = (acc[nt][rt][r] + bias) * sWgt[row];
                atomicAdd(&out[(size_t)sTok[row] * HD + hcol], val);
            }
        }
    }
}

// ---------------- host launcher ----------------
extern "C" void kernel_launch(void* const* d_in, const int* in_sizes, int n_in,
                              void* d_out, int out_size, void* d_ws, size_t ws_size,
                              hipStream_t stream) {
    const float* x   = (const float*)d_in[0];   // [B,S,H]
    const float* Wg  = (const float*)d_in[1];   // [H,E]
    const float* w1  = (const float*)d_in[2];   // [E,H,4H]
    const float* b1  = (const float*)d_in[3];   // [E,4H]
    const float* w2  = (const float*)d_in[4];   // [E,4H,H]
    const float* b2  = (const float*)d_in[5];   // [E,H]
    float* out = (float*)d_out;

    // workspace layout (bytes)
    char* ws = (char*)d_ws;
    const size_t off_counts = 0;                                   // 256 B (uses 32)
    const size_t off_tok    = 256;                                 // E*MAXT*4 = 512 KB
    const size_t off_wgt    = off_tok + (size_t)ED * MAXT * 4;     // 512 KB
    const size_t off_x      = off_wgt + (size_t)ED * MAXT * 4;     // NTOK*HD*2 = 16 MB
    const size_t off_w1t    = off_x   + (size_t)NTOK * HD * 2;     // E*FH*HD*2 = 16 MB
    const size_t off_w2t    = off_w1t + (size_t)ED * FH * HD * 2;  // 16 MB
    int*    counts = (int*)(ws + off_counts);
    int*    tok    = (int*)(ws + off_tok);
    float*  wgt    = (float*)(ws + off_wgt);
    __bf16* xbf    = (__bf16*)(ws + off_x);
    __bf16* w1t    = (__bf16*)(ws + off_w1t);
    __bf16* w2t    = (__bf16*)(ws + off_w2t);

    // fresh state every call (graph replays must be deterministic)
    hipMemsetAsync(d_out, 0, (size_t)out_size * sizeof(float), stream);
    hipMemsetAsync(ws + off_counts, 0, 256, stream);

    // 1) precision conversion + weight transposes (L2-resident after first touch)
    {
        int n = NTOK * HD;
        cvt_bf16_kernel<<<(n + 255) / 256, 256, 0, stream>>>(x, xbf, n);
    }
    transpose_cvt_kernel<<<dim3(FH / 32, HD / 32, ED), 256, 0, stream>>>(w1, w1t, HD, FH);
    transpose_cvt_kernel<<<dim3(HD / 32, FH / 32, ED), 256, 0, stream>>>(w2, w2t, FH, HD);

    // 2) gating + bucket build
    gating_kernel<<<NTOK / 256, 256, 0, stream>>>(x, Wg, counts, tok, wgt);

    // 3) fused expert FFN over routed tiles
    moe_expert_kernel<<<dim3(MAX_TILES, ED), 256, 0, stream>>>(
        xbf, w1t, w2t, b1, b2, counts, tok, wgt, out);
}
